// MultiHeadAttentionWithRoPE_41927470743828
// MI455X (gfx1250) — compile-verified, matrix operand order checked by reading, not run on hardware
//
#include <hip/hip_runtime.h>
#include <math.h>

// ---------------------------------------------------------------------------
// MultiHeadAttention + RoPE for MI455X (gfx1250).
// All matmuls: v_wmma_f32_16x16x32_bf16 (fp32 accumulate).
// Global->LDS staging: double-buffered GLOBAL_LOAD_ASYNC_TO_LDS_B128
// (ASYNCcnt path), sync fallback otherwise.
// B=4 S=2048 E=1024 H=16 D=64.
// ---------------------------------------------------------------------------

typedef __bf16 bf16;
typedef __bf16 v16bf __attribute__((ext_vector_type(16)));
typedef __bf16 v8bf  __attribute__((ext_vector_type(8)));
typedef __bf16 v4bf  __attribute__((ext_vector_type(4)));
typedef float v8f    __attribute__((ext_vector_type(8)));
typedef int   v4i    __attribute__((vector_size(16)));   // builtin's pointee type

constexpr int Bb = 4;
constexpr int Ss = 2048;
constexpr int Ee = 1024;
constexpr int Hh = 16;
constexpr int Dd = 64;
constexpr int BS = Bb * Ss;   // 8192

#if defined(__has_builtin)
#if __has_builtin(__builtin_amdgcn_global_load_async_to_lds_b128)
#define HAVE_ASYNC_LDS 1
#endif
#endif

#define AS1 __attribute__((address_space(1)))
#define AS3 __attribute__((address_space(3)))

// 16-byte global -> LDS copy. Async (no VGPR staging) when available.
__device__ inline void cp16(bf16* ldst, const bf16* gsrc) {
#ifdef HAVE_ASYNC_LDS
    __builtin_amdgcn_global_load_async_to_lds_b128(
        (AS1 v4i*)(uintptr_t)gsrc,
        (AS3 v4i*)(uint32_t)(uintptr_t)ldst,
        0, 0);
#else
    *(v8bf*)ldst = *(const v8bf*)gsrc;
#endif
}

__device__ inline void async_wait_all() {
#ifdef HAVE_ASYNC_LDS
#if __has_builtin(__builtin_amdgcn_s_wait_asynccnt)
    __builtin_amdgcn_s_wait_asynccnt(0);
#else
    asm volatile("s_wait_asynccnt 0" ::: "memory");
#endif
#endif
}

__device__ inline v8f vzero8() {
    v8f z;
#pragma unroll
    for (int r = 0; r < 8; ++r) z[r] = 0.0f;
    return z;
}

// Load a 16-bit WMMA fragment (A or B role) from a row-major LDS slab.
// Per ISA 7.12.2: VGPR0-3 = K[kbase+kh*8..+7], VGPR4-7 = K[kbase+16+kh*8..+7]
__device__ inline v16bf ld_frag(const bf16* rowbase, int kbase, int lane) {
    const int kh = (lane >> 4) & 1;
    union { v16bf v; v8bf h[2]; } u;
    u.h[0] = *(const v8bf*)(rowbase + kbase + kh * 8);
    u.h[1] = *(const v8bf*)(rowbase + kbase + 16 + kh * 8);
    return u.v;
}

__device__ inline float rmax16(float v) {
    v = fmaxf(v, __shfl_xor(v, 1, 32));
    v = fmaxf(v, __shfl_xor(v, 2, 32));
    v = fmaxf(v, __shfl_xor(v, 4, 32));
    v = fmaxf(v, __shfl_xor(v, 8, 32));
    return v;
}
__device__ inline float rsum16(float v) {
    v += __shfl_xor(v, 1, 32);
    v += __shfl_xor(v, 2, 32);
    v += __shfl_xor(v, 4, 32);
    v += __shfl_xor(v, 8, 32);
    return v;
}

// -------------------------- fp32 -> bf16 convert ---------------------------
__global__ void cvt_kernel(const float* __restrict__ in, bf16* __restrict__ out, int n4) {
    int i = blockIdx.x * 256 + threadIdx.x;
    if (i < n4) {
        float4 f = ((const float4*)in)[i];
        v4bf o;
        o[0] = (bf16)f.x; o[1] = (bf16)f.y; o[2] = (bf16)f.z; o[3] = (bf16)f.w;
        ((v4bf*)out)[i] = o;
    }
}

// --------------- transpose + convert: W[K,N] fp32 -> Wt[N,K] bf16 ----------
__global__ __launch_bounds__(256) void transpose_cvt_kernel(const float* __restrict__ in,
                                                            bf16* __restrict__ out) {
    __shared__ float tile[32][33];
    const int bx = blockIdx.x * 32;   // n
    const int by = blockIdx.y * 32;   // k
    const int tx = threadIdx.x & 31;
    const int ty = threadIdx.x >> 5;  // 0..7
#pragma unroll
    for (int r = 0; r < 32; r += 8)
        tile[ty + r][tx] = in[(size_t)(by + ty + r) * Ee + bx + tx];
    __syncthreads();
#pragma unroll
    for (int r = 0; r < 32; r += 8)
        out[(size_t)(bx + ty + r) * Ee + by + tx] = (bf16)tile[tx][ty + r];
}

// ------------------------------ bf16 GEMM ----------------------------------
// C[M,N] = A[M,K] * Wt[N,K]^T + bias.  Block tile 128x64, 8 waves, each wave
// 32x32 = 2x2 WMMA accumulators, K-step 32, double-buffered async LDS tiles.
// OUT_MODE: 0 = fp32 [M,N]; 1 = bf16 [M,N]; 2 = bf16 transposed vT[b,h,d,s].
template <int OUT_MODE>
__global__ __launch_bounds__(256) void gemm_kernel(const bf16* __restrict__ A,
                                                   const bf16* __restrict__ Wt,
                                                   const float* __restrict__ bias,
                                                   bf16* __restrict__ outb,
                                                   float* __restrict__ outf,
                                                   int M, int N, int K) {
    __shared__ __align__(16) bf16 ldsA[2][128 * 32];
    __shared__ __align__(16) bf16 ldsB[2][64 * 32];

    const int tid = threadIdx.x;
    const int lane = tid & 31;
    const int wave = tid >> 5;
    const int lr = lane & 15;
    const int m0 = blockIdx.y * 128;
    const int n0 = blockIdx.x * 64;
    const int wm = (wave >> 1) * 32;  // 0,32,64,96
    const int wn = (wave & 1) * 32;   // 0,32

    auto stage = [&](int kb, int buf) {
#pragma unroll
        for (int i = 0; i < 2; ++i) {
            const int c = tid + i * 256;          // 512 chunks of 16B (A)
            const int r = c >> 2, seg = c & 3;
            cp16(ldsA[buf] + r * 32 + seg * 8,
                 A + (size_t)(m0 + r) * K + kb + seg * 8);
        }
        {
            const int r = tid >> 2, seg = tid & 3; // 256 chunks (B)
            cp16(ldsB[buf] + r * 32 + seg * 8,
                 Wt + (size_t)(n0 + r) * K + kb + seg * 8);
        }
    };

    v8f acc[2][2];
#pragma unroll
    for (int i = 0; i < 2; ++i)
#pragma unroll
        for (int j = 0; j < 2; ++j) acc[i][j] = vzero8();

    stage(0, 0);
    async_wait_all();
    __syncthreads();

    int cur = 0;
    for (int kb = 0; kb < K; kb += 32) {
        if (kb + 32 < K) stage(kb + 32, cur ^ 1);   // prefetch next tile

        v16bf bfr[2];
#pragma unroll
        for (int j = 0; j < 2; ++j)
            bfr[j] = ld_frag(ldsB[cur] + (wn + j * 16 + lr) * 32, 0, lane);
#pragma unroll
        for (int i = 0; i < 2; ++i) {
            v16bf a = ld_frag(ldsA[cur] + (wm + i * 16 + lr) * 32, 0, lane);
#pragma unroll
            for (int j = 0; j < 2; ++j)
                acc[i][j] = __builtin_amdgcn_wmma_f32_16x16x32_bf16(
                    false, a, false, bfr[j], (short)0, acc[i][j], false, false);
        }
        async_wait_all();
        __syncthreads();
        cur ^= 1;
    }

    // Epilogue: C/D layout -> lane holds col n = lane&15, rows r + 8*(lane>>4)
    const int mg = lane >> 4;
#pragma unroll
    for (int i = 0; i < 2; ++i) {
#pragma unroll
        for (int j = 0; j < 2; ++j) {
            const int ncol = n0 + wn + j * 16 + lr;
            const float bv = bias[ncol];
            if (OUT_MODE == 2) {
                // pack 8 consecutive s-rows, single 16B store into vT[b,h,d,s]
                const int mrow0 = m0 + wm + i * 16 + 8 * mg;
                const int bidx = mrow0 >> 11;         // /Ss
                const int s0 = mrow0 & (Ss - 1);
                v8bf pk;
#pragma unroll
                for (int r = 0; r < 8; ++r) pk[r] = (bf16)(acc[i][j][r] + bv);
                const size_t dst =
                    ((size_t)(bidx * Hh + (ncol >> 6)) * Dd + (ncol & 63)) * Ss + s0;
                *(v8bf*)(outb + dst) = pk;
            } else {
#pragma unroll
                for (int r = 0; r < 8; ++r) {
                    const int mrow = m0 + wm + i * 16 + r + 8 * mg;
                    const float v = acc[i][j][r] + bv;
                    if (OUT_MODE == 1)
                        outb[(size_t)mrow * N + ncol] = (bf16)v;
                    else
                        outf[(size_t)mrow * N + ncol] = v;
                }
            }
        }
    }
}

// ------------------------------ RoPE (in place) ----------------------------
__global__ void rope_kernel(bf16* __restrict__ t) {
    const int idx = blockIdx.x * 256 + threadIdx.x;  // B*S*H*32 total
    const int i = idx & 31;
    const int h = (idx >> 5) & (Hh - 1);
    const int s = (idx >> 9) & (Ss - 1);
    const int b = idx >> 20;
    const size_t base = ((size_t)(b * Ss + s)) * Ee + h * 64;
    const float x1 = (float)t[base + i];
    const float x2 = (float)t[base + i + 32];
    // theta_i = 10000^(-i/32); ln(10000)/32 = 0.28782313662425572
    const float ang = (float)s * __expf(-(float)i * 0.28782313662425572f);
    float sn, cs;
    __sincosf(ang, &sn, &cs);
    t[base + i]      = (bf16)(x1 * cs - x2 * sn);
    t[base + i + 32] = (bf16)(x2 * cs + x1 * sn);
}

// --------------------------- flash attention -------------------------------
// grid: (S/128, B*H), 256 threads = 8 waves; wave owns 16 query rows.
// Double-buffered async K [t][d] and Vt [d][t] tiles (64 keys / iteration);
// online softmax; P converted C-layout -> A-layout through per-wave LDS.
__global__ __launch_bounds__(256) void attn_kernel(const bf16* __restrict__ qb,
                                                   const bf16* __restrict__ kbuf,
                                                   const bf16* __restrict__ vT,
                                                   const int* __restrict__ mask,
                                                   bf16* __restrict__ outb) {
    __shared__ __align__(16) bf16 ldsK[2][64 * 64];   // [key][d]
    __shared__ __align__(16) bf16 ldsVt[2][64 * 64];  // [d][key]
    __shared__ __align__(16) bf16 ldsP[8][16 * 64];

    const int tid = threadIdx.x;
    const int lane = tid & 31;
    const int wave = tid >> 5;
    const int lr = lane & 15;
    const int kh = lane >> 4;
    const int bh = blockIdx.y;
    const int b = bh >> 4;
    const int h = bh & 15;
    const int q0 = blockIdx.x * 128 + wave * 16;

    const bf16* kbase = kbuf + (size_t)b * Ss * Ee + h * 64;     // + t*Ee
    const bf16* vtbase = vT + ((size_t)(b * Hh + h)) * Dd * Ss;  // + d*Ss + t

    auto stage = [&](int t0, int buf) {
#pragma unroll
        for (int i = 0; i < 2; ++i) {
            const int c = tid + i * 256;    // 512 chunks of 16B
            const int t = c >> 3, dseg = c & 7;
            cp16(ldsK[buf] + t * 64 + dseg * 8,
                 kbase + (size_t)(t0 + t) * Ee + dseg * 8);
        }
#pragma unroll
        for (int i = 0; i < 2; ++i) {
            const int c = tid + i * 256;
            const int d = c >> 3, tseg = c & 7;
            cp16(ldsVt[buf] + d * 64 + tseg * 8,
                 vtbase + (size_t)d * Ss + t0 + tseg * 8);
        }
    };

    // Q fragments (16 rows x 64 d), kept in registers for the whole kernel
    const bf16* qrow = qb + ((size_t)(b * Ss + q0 + lr)) * Ee + h * 64;
    v16bf qf[2];
#pragma unroll
    for (int ks = 0; ks < 2; ++ks) {
        union { v16bf v; v8bf hh[2]; } u;
        u.hh[0] = *(const v8bf*)(qrow + ks * 32 + kh * 8);
        u.hh[1] = *(const v8bf*)(qrow + ks * 32 + 16 + kh * 8);
        qf[ks] = u.v;
    }

    float mrow[8], lsum[8];
    v8f o[4];
#pragma unroll
    for (int r = 0; r < 8; ++r) { mrow[r] = -3.0e38f; lsum[r] = 0.0f; }
#pragma unroll
    for (int dt = 0; dt < 4; ++dt) o[dt] = vzero8();

    stage(0, 0);
    async_wait_all();
    __syncthreads();

    int cur = 0;
    for (int t0 = 0; t0 < Ss; t0 += 64) {
        if (t0 + 64 < Ss) stage(t0 + 64, cur ^ 1);   // prefetch next key block

        // scores: 4 tiles of 16q x 16k, contraction over D=64 (2 WMMA each)
        float p[4][8];
#pragma unroll
        for (int j = 0; j < 4; ++j) {
            v8f st = vzero8();
            const bf16* krow = ldsK[cur] + (16 * j + lr) * 64;
            v16bf kf0 = ld_frag(krow, 0, lane);
            v16bf kf1 = ld_frag(krow, 32, lane);
            st = __builtin_amdgcn_wmma_f32_16x16x32_bf16(false, qf[0], false, kf0,
                                                         (short)0, st, false, false);
            st = __builtin_amdgcn_wmma_f32_16x16x32_bf16(false, qf[1], false, kf1,
                                                         (short)0, st, false, false);
            // additive mask: 0 where mask==1, -1e9 where mask==0
            const float madd =
                mask[b * Ss + t0 + 16 * j + lr] ? 0.0f : -1.0e9f;
#pragma unroll
            for (int r = 0; r < 8; ++r)
                p[j][r] = fmaf(st[r], 0.125f, madd);
        }

        // online softmax (rows spread across 16 lanes of each half-wave)
        float alpha[8];
#pragma unroll
        for (int r = 0; r < 8; ++r) {
            float mx = fmaxf(fmaxf(p[0][r], p[1][r]), fmaxf(p[2][r], p[3][r]));
            mx = rmax16(mx);
            const float mnew = fmaxf(mrow[r], mx);
            alpha[r] = __expf(mrow[r] - mnew);
            mrow[r] = mnew;
            float rs = 0.0f;
#pragma unroll
            for (int j = 0; j < 4; ++j) {
                p[j][r] = __expf(p[j][r] - mnew);
                rs += p[j][r];
            }
            rs = rsum16(rs);
            lsum[r] = lsum[r] * alpha[r] + rs;
        }
#pragma unroll
        for (int dt = 0; dt < 4; ++dt)
#pragma unroll
            for (int r = 0; r < 8; ++r) o[dt][r] *= alpha[r];

        // P (C-layout) -> per-wave LDS -> A-layout fragments
        bf16* pp = ldsP[wave];
#pragma unroll
        for (int j = 0; j < 4; ++j)
#pragma unroll
            for (int r = 0; r < 8; ++r)
                pp[(r + 8 * kh) * 64 + 16 * j + lr] = (bf16)p[j][r];

#pragma unroll
        for (int ks = 0; ks < 2; ++ks) {
            v16bf pf = ld_frag(pp + lr * 64, ks * 32, lane);
#pragma unroll
            for (int dt = 0; dt < 4; ++dt) {
                v16bf vf = ld_frag(ldsVt[cur] + (16 * dt + lr) * 64, ks * 32, lane);
                o[dt] = __builtin_amdgcn_wmma_f32_16x16x32_bf16(
                    false, pf, false, vf, (short)0, o[dt], false, false);
            }
        }

        async_wait_all();
        __syncthreads();
        cur ^= 1;
    }

    // normalize and write attn output in [B,S,E] bf16
#pragma unroll
    for (int r = 0; r < 8; ++r) {
        const float inv = 1.0f / lsum[r];
        const size_t rowbase = ((size_t)(b * Ss + q0 + r + 8 * kh)) * Ee + h * 64;
#pragma unroll
        for (int dt = 0; dt < 4; ++dt)
            outb[rowbase + 16 * dt + lr] = (bf16)(o[dt][r] * inv);
    }
}

// ------------------------------ launcher -----------------------------------
extern "C" void kernel_launch(void* const* d_in, const int* in_sizes, int n_in,
                              void* d_out, int out_size, void* d_ws, size_t ws_size,
                              hipStream_t stream) {
    const float* x  = (const float*)d_in[0];
    const int* attn_mask = (const int*)d_in[1];
    const float* Wq = (const float*)d_in[2];
    const float* bq = (const float*)d_in[3];
    const float* Wk = (const float*)d_in[4];
    const float* bk = (const float*)d_in[5];
    const float* Wv = (const float*)d_in[6];
    const float* bv = (const float*)d_in[7];
    const float* Wo = (const float*)d_in[8];
    const float* bo = (const float*)d_in[9];
    float* out = (float*)d_out;

    char* w = (char*)d_ws;
    const size_t actB = (size_t)BS * Ee * sizeof(bf16);  // 16 MB
    const size_t wtB  = (size_t)Ee * Ee * sizeof(bf16);  //  2 MB
    bf16* xb    = (bf16*)w; w += actB;
    bf16* WqT   = (bf16*)w; w += wtB;
    bf16* WkT   = (bf16*)w; w += wtB;
    bf16* WvT   = (bf16*)w; w += wtB;
    bf16* WoT   = (bf16*)w; w += wtB;
    bf16* qbuf  = (bf16*)w; w += actB;
    bf16* kbuf  = (bf16*)w; w += actB;
    bf16* vTbuf = (bf16*)w; w += actB;   // V transposed: [B,H,D,S]
    bf16* attnb = (bf16*)w; w += actB;   // total 88 MB

    // 1) convert activations, transpose-convert weights
    cvt_kernel<<<(BS * Ee / 4 + 255) / 256, 256, 0, stream>>>(x, xb, BS * Ee / 4);
    dim3 tg(Ee / 32, Ee / 32);
    transpose_cvt_kernel<<<tg, 256, 0, stream>>>(Wq, WqT);
    transpose_cvt_kernel<<<tg, 256, 0, stream>>>(Wk, WkT);
    transpose_cvt_kernel<<<tg, 256, 0, stream>>>(Wv, WvT);
    transpose_cvt_kernel<<<tg, 256, 0, stream>>>(Wo, WoT);

    // 2) QKV projections (Q,K -> bf16 [B,S,E]; V -> bf16 transposed [B,H,D,S])
    dim3 gg(Ee / 64, BS / 128);
    gemm_kernel<1><<<gg, 256, 0, stream>>>(xb, WqT, bq, qbuf, nullptr, BS, Ee, Ee);
    gemm_kernel<1><<<gg, 256, 0, stream>>>(xb, WkT, bk, kbuf, nullptr, BS, Ee, Ee);
    gemm_kernel<2><<<gg, 256, 0, stream>>>(xb, WvT, bv, vTbuf, nullptr, BS, Ee, Ee);

    // 3) RoPE on Q and K
    const int ropeN = Bb * Ss * Hh * 32;
    rope_kernel<<<ropeN / 256, 256, 0, stream>>>(qbuf);
    rope_kernel<<<ropeN / 256, 256, 0, stream>>>(kbuf);

    // 4) attention
    dim3 ag(Ss / 128, Bb * Hh);
    attn_kernel<<<ag, 256, 0, stream>>>(qbuf, kbuf, vTbuf, attn_mask, attnb);

    // 5) output projection (fp32 out)
    gemm_kernel<0><<<gg, 256, 0, stream>>>(attnb, WoT, bo, nullptr, out, BS, Ee, Ee);
}
